// LSTMModel_24584392802526
// MI455X (gfx1250) — matrix operand order repaired and checked by reading, not compiled, not run on hardware
//
#include <hip/hip_runtime.h>
#include <hip/hip_bf16.h>

// ---------------- problem constants ----------------
#define H        512
#define IN_DIM   64
#define STEPS    2048
#define NB       128        // persistent workgroups
#define UPW      4          // hidden units owned per workgroup
// per-WG bf16 weight slice layout (element offsets):
//   [Whh0: 16x512][Wih0: 16x64][Wih1: 16x512][Whh1: 16x512]  rows r = gate*4 + unit
#define SLICE    25600
#define W_WHH0   0
#define W_WIH0   8192
#define W_WIH1   9216
#define W_WHH1   17408

typedef __attribute__((ext_vector_type(16))) __bf16 v16bf;
typedef __attribute__((ext_vector_type(8)))  __bf16 v8bf;
typedef __attribute__((ext_vector_type(8)))  float  v8f;
typedef __attribute__((ext_vector_type(4)))  unsigned int u32x4;
typedef __attribute__((ext_vector_type(8)))  int i32x8;
typedef __attribute__((ext_vector_type(4)))  int i32x4;

__device__ __forceinline__ unsigned short f2bf(float f) {
  unsigned u = __float_as_uint(f);
  u += 0x7FFFu + ((u >> 16) & 1u);          // round-to-nearest-even (sans NaN care)
  return (unsigned short)(u >> 16);
}
__device__ __forceinline__ float sigm(float x) { return 1.0f / (1.0f + __expf(-x)); }

// ---------------- prep: LayerNorm'd sliding windows -> bf16 x[t][64] ----------------
__global__ void prep_x(const float* __restrict__ batch, const float* __restrict__ ln_w,
                       const float* __restrict__ ln_b, unsigned short* __restrict__ x_bf) {
  const int t = blockIdx.x, i = threadIdx.x;   // 64 threads
  __shared__ float sv[64], sq[64];
  const int idx = t - (IN_DIM - 1) + i;
  const float v = (idx >= 0) ? batch[idx] : 0.f;
  sv[i] = v; sq[i] = v * v;
  __syncthreads();
  for (int s = 32; s > 0; s >>= 1) {
    if (i < s) { sv[i] += sv[i + s]; sq[i] += sq[i + s]; }
    __syncthreads();
  }
  const float mu  = sv[0] * (1.0f / IN_DIM);
  const float var = sq[0] * (1.0f / IN_DIM) - mu * mu;
  const float x   = (v - mu) * rsqrtf(var + 1e-5f) * ln_w[i] + ln_b[i];
  x_bf[t * IN_DIM + i] = f2bf(x);
}

// ---------------- prep: per-WG bf16 weight slices + combined biases ----------------
__global__ void prep_w(const float* __restrict__ Wih0, const float* __restrict__ Whh0,
                       const float* __restrict__ bih0, const float* __restrict__ bhh0,
                       const float* __restrict__ Wih1, const float* __restrict__ Whh1,
                       const float* __restrict__ bih1, const float* __restrict__ bhh1,
                       unsigned short* __restrict__ wslice, float* __restrict__ bias) {
  const int wg = blockIdx.x;
  unsigned short* dst = wslice + (size_t)wg * SLICE;
  for (int e = threadIdx.x; e < SLICE; e += 256) {
    float v; int r, k, grow;
    if (e < W_WIH0) {                 r = e >> 9;            k = e & 511;
      grow = (r >> 2) * H + wg * UPW + (r & 3); v = Whh0[grow * H + k];
    } else if (e < W_WIH1) { int e2 = e - W_WIH0; r = e2 >> 6; k = e2 & 63;
      grow = (r >> 2) * H + wg * UPW + (r & 3); v = Wih0[grow * IN_DIM + k];
    } else if (e < W_WHH1) { int e2 = e - W_WIH1; r = e2 >> 9; k = e2 & 511;
      grow = (r >> 2) * H + wg * UPW + (r & 3); v = Wih1[grow * H + k];
    } else {                 int e2 = e - W_WHH1; r = e2 >> 9; k = e2 & 511;
      grow = (r >> 2) * H + wg * UPW + (r & 3); v = Whh1[grow * H + k];
    }
    dst[e] = f2bf(v);
  }
  if (threadIdx.x < 32) {
    const int layer = threadIdx.x >> 4, r = threadIdx.x & 15;
    const int grow = (r >> 2) * H + wg * UPW + (r & 3);
    bias[wg * 32 + threadIdx.x] =
        (layer == 0) ? (bih0[grow] + bhh0[grow]) : (bih1[grow] + bhh1[grow]);
  }
}

// ---------------- prep: zero hidden state + grid barrier ----------------
__global__ void init_state(unsigned short* h0, unsigned short* h1, unsigned* bar) {
  for (int j = threadIdx.x; j < H; j += 256) { h0[j] = 0; h1[j] = 0; }
  if (threadIdx.x < 2) bar[threadIdx.x] = 0u;
}

// ---------------- WMMA mat-vec core ----------------
// A tile = 16 owned gate-rows x 32 K-columns (bf16 in LDS, row-major, stride `stride`).
// B tile = hidden/input vector broadcast over 16 columns: B[k][n] = bvec[k].
// D column 0 == W*h (all columns identical). K split 4-ways across waves (kq = wave&3).
// `wave_u`/`kq` are SGPR values (readfirstlane) so all branches here are scalar and
// EXEC stays all-ones across every WMMA (ISA requirement).
template <int NCHUNKS>
__device__ __forceinline__ void gate_matvec(const unsigned short* __restrict__ w_lds,
                                            int wbase, int stride,
                                            const unsigned short* __restrict__ bvec,
                                            int wave_u, int lane, float (*partial)[16]) {
  const int kq   = wave_u & 3;             // scalar
  const int row  = lane & 15;
  const int hi   = (lane & 16) ? 16 : 0;   // A layout: lanes 16-31 hold K=16..31 of chunk
  const int koff = (lane & 16) ? 16 : 0;   // B layout: lanes 16-31 hold K=16..31 of chunk
  const int rowoff = wbase + row * stride;
  v8f acc = {0.f, 0.f, 0.f, 0.f, 0.f, 0.f, 0.f, 0.f};
  if (NCHUNKS >= 4) {
#pragma unroll
    for (int j = 0; j < NCHUNKS / 4; ++j) {
      const int cbase = (kq + 4 * j) * 32;
      v8bf a0 = *(const v8bf*)(w_lds + rowoff + cbase + hi);
      v8bf a1 = *(const v8bf*)(w_lds + rowoff + cbase + 8 + hi);
      v16bf a;
#pragma unroll
      for (int e = 0; e < 8; ++e) { a[e] = a0[e]; a[e + 8] = a1[e]; }
      v16bf b = *(const v16bf*)(bvec + cbase + koff);
      acc = __builtin_amdgcn_wmma_f32_16x16x32_bf16(false, a, false, b, (short)0, acc,
                                                    false, false);
    }
  } else {
    if (kq < NCHUNKS) {                    // scalar branch, EXEC untouched
      const int cbase = kq * 32;
      v8bf a0 = *(const v8bf*)(w_lds + rowoff + cbase + hi);
      v8bf a1 = *(const v8bf*)(w_lds + rowoff + cbase + 8 + hi);
      v16bf a;
#pragma unroll
      for (int e = 0; e < 8; ++e) { a[e] = a0[e]; a[e + 8] = a1[e]; }
      v16bf b = *(const v16bf*)(bvec + cbase + koff);
      acc = __builtin_amdgcn_wmma_f32_16x16x32_bf16(false, a, false, b, (short)0, acc,
                                                    false, false);
    }
  }
  // extract column 0: lane0 holds rows 0..7, lane16 rows 8..15
  if ((lane & 15) == 0) {
    const int base = (lane >> 4) * 8;
#pragma unroll
    for (int r = 0; r < 8; ++r) partial[wave_u][base + r] = acc[r];
  }
}

// ---------------- grid-wide sense barrier ----------------
__device__ __forceinline__ void grid_barrier(unsigned* bar, unsigned& local_phase) {
  __threadfence();
  __syncthreads();
  if (threadIdx.x == 0) {
    const unsigned ph = local_phase;
    const unsigned arr = atomicAdd(&bar[0], 1u);
    if (arr == NB - 1) {
      atomicExch(&bar[0], 0u);
      __threadfence();
      atomicAdd(&bar[1], 1u);
    } else {
      while (atomicAdd(&bar[1], 0u) == ph) { __builtin_amdgcn_s_sleep(8); }
    }
  }
  local_phase++;
  __syncthreads();
  __threadfence();
}

// ---------------- persistent LSTM kernel ----------------
__global__ void __launch_bounds__(256, 1)
lstm_persist(const unsigned short* __restrict__ wslice_g, const float* __restrict__ bias_g,
             const unsigned short* __restrict__ x_bf, unsigned short* h0g,
             unsigned short* h1g, float* __restrict__ h1hist, unsigned* bar) {
  __shared__ __align__(32) unsigned short w_lds[SLICE];  // 51200 B bf16 weights
  __shared__ __align__(32) unsigned short hA[H];
  __shared__ __align__(32) unsigned short hB[H];
  __shared__ __align__(32) unsigned short xw[IN_DIM];
  __shared__ float partial[8][16];
  __shared__ float gsum[16];
  __shared__ float bias_l[2][16];
  __shared__ float cstate[2][UPW];

  const int tid = threadIdx.x, lane = tid & 31;
  const int wave_u = __builtin_amdgcn_readfirstlane(tid >> 5);  // scalar wave id
  const int wg = blockIdx.x;

  // ---- stage this WG's weight slice LDS <- global via the Tensor Data Mover ----
  if (wave_u == 0) {
    const unsigned lds_addr = (unsigned)(unsigned long long)(const void*)w_lds;
    const unsigned long long ga =
        (unsigned long long)(const void*)(wslice_g + (size_t)wg * SLICE);
    const unsigned ndw = SLICE / 2;                    // 12800 dwords = 51200 B
    u32x4 g0;
    g0[0] = 1u;                                        // count = 1 (valid D#)
    g0[1] = lds_addr;                                  // lds_addr
    g0[2] = (unsigned)(ga & 0xFFFFFFFFull);            // global_addr[31:0]
    g0[3] = (unsigned)((ga >> 32) & 0x01FFFFFFull) | (2u << 30);  // addr[56:32] | type=2
    i32x8 g1;
    g1[0] = (int)(2u << 16);                           // data_size = 4B, no multicast
    g1[1] = (int)((ndw & 0xFFFFu) << 16);              // tensor_dim0[15:0]
    g1[2] = (int)((ndw >> 16) | (1u << 16));           // tensor_dim0[31:16] | tensor_dim1=1
    g1[3] = (int)((ndw & 0xFFFFu) << 16);              // tile_dim0 = ndw
    g1[4] = 1;                                         // tile_dim1 = 1, tile_dim2 = 0
    g1[5] = (int)ndw;                                  // tensor_dim0_stride lo32
    g1[6] = 0;
    g1[7] = 0;
    i32x4 z4 = {0, 0, 0, 0};                           // D# groups 2/3 unused (<=2D)
    i32x8 z8 = {0, 0, 0, 0, 0, 0, 0, 0};
    __builtin_amdgcn_tensor_load_to_lds(g0, g1, z4, z4, z8, 0);
    __builtin_amdgcn_s_wait_tensorcnt(0);
  }
  if (tid < 32) bias_l[tid >> 4][tid & 15] = bias_g[wg * 32 + tid];
  if (tid < 2 * UPW) cstate[tid >> 2][tid & 3] = 0.f;
  __syncthreads();

  unsigned local_phase = 0;
  for (int t = 0; t < STEPS; ++t) {
    // ---------- phase A: layer 0 ----------
    for (int i = tid; i < H; i += 256) hA[i] = ((volatile const unsigned short*)h0g)[i];
    if (tid < IN_DIM) xw[tid] = x_bf[t * IN_DIM + tid];
    __syncthreads();
    if (wave_u < 4) gate_matvec<16>(w_lds, W_WHH0, H,      hA, wave_u, lane, partial);
    else            gate_matvec<2>(w_lds, W_WIH0, IN_DIM,  xw, wave_u, lane, partial);
    __syncthreads();
    if (tid < 16) {
      float s = bias_l[0][tid];
#pragma unroll
      for (int w = 0; w < 8; ++w) s += partial[w][tid];
      gsum[tid] = s;
    }
    __syncthreads();
    if (tid < UPW) {
      const float ig = sigm(gsum[0 * UPW + tid]);
      const float fg = sigm(gsum[1 * UPW + tid]);
      const float gg = tanhf(gsum[2 * UPW + tid]);
      const float og = sigm(gsum[3 * UPW + tid]);
      const float c  = fg * cstate[0][tid] + ig * gg;
      cstate[0][tid] = c;
      h0g[wg * UPW + tid] = f2bf(og * tanhf(c));
    }
    grid_barrier(bar, local_phase);

    // ---------- phase B: layer 1 ----------
    for (int i = tid; i < H; i += 256) {
      hA[i] = ((volatile const unsigned short*)h0g)[i];
      hB[i] = ((volatile const unsigned short*)h1g)[i];
    }
    __syncthreads();
    if (wave_u < 4) gate_matvec<16>(w_lds, W_WHH1, H, hB, wave_u, lane, partial);
    else            gate_matvec<16>(w_lds, W_WIH1, H, hA, wave_u, lane, partial);
    __syncthreads();
    if (tid < 16) {
      float s = bias_l[1][tid];
#pragma unroll
      for (int w = 0; w < 8; ++w) s += partial[w][tid];
      gsum[tid] = s;
    }
    __syncthreads();
    if (tid < UPW) {
      const float ig = sigm(gsum[0 * UPW + tid]);
      const float fg = sigm(gsum[1 * UPW + tid]);
      const float gg = tanhf(gsum[2 * UPW + tid]);
      const float og = sigm(gsum[3 * UPW + tid]);
      const float c  = fg * cstate[1][tid] + ig * gg;
      cstate[1][tid] = c;
      const float h  = og * tanhf(c);
      h1g[wg * UPW + tid] = f2bf(h);
      h1hist[(size_t)t * H + wg * UPW + tid] = h;
    }
    grid_barrier(bar, local_phase);
  }
}

// ---------------- final projection: out[t] = relu(h1(t)) . fc_w + fc_b ----------------
__global__ void out_proj(const float* __restrict__ h1hist, const float* __restrict__ fc_w,
                         const float* __restrict__ fc_b, float* __restrict__ out) {
  const int t = blockIdx.x, tid = threadIdx.x;  // 256 threads
  __shared__ float red[256];
  float s = 0.f;
  for (int j = tid; j < H; j += 256) {
    float h = h1hist[(size_t)t * H + j];
    h = h > 0.f ? h : 0.f;
    s += h * fc_w[j];
  }
  red[tid] = s;
  __syncthreads();
  for (int st = 128; st > 0; st >>= 1) {
    if (tid < st) red[tid] += red[tid + st];
    __syncthreads();
  }
  if (tid == 0) out[t] = red[0] + fc_b[0];
}

// ---------------- launch ----------------
extern "C" void kernel_launch(void* const* d_in, const int* in_sizes, int n_in,
                              void* d_out, int out_size, void* d_ws, size_t ws_size,
                              hipStream_t stream) {
  const float* batch = (const float*)d_in[0];
  const float* Wih0  = (const float*)d_in[1];
  const float* Whh0  = (const float*)d_in[2];
  const float* bih0  = (const float*)d_in[3];
  const float* bhh0  = (const float*)d_in[4];
  const float* Wih1  = (const float*)d_in[5];   // [1, 4H, H]
  const float* Whh1  = (const float*)d_in[6];
  const float* bih1  = (const float*)d_in[7];
  const float* bhh1  = (const float*)d_in[8];
  const float* ln_w  = (const float*)d_in[9];
  const float* ln_b  = (const float*)d_in[10];
  const float* fc_w  = (const float*)d_in[11];
  const float* fc_b  = (const float*)d_in[12];
  float* out = (float*)d_out;

  unsigned char* ws = (unsigned char*)d_ws;
  unsigned short* x_bf = (unsigned short*)ws;                          // 262144 B
  unsigned short* wsl  = (unsigned short*)(ws + 262144);               // 6553600 B
  float*          bias = (float*)(ws + 6815744);                       // 16384 B
  unsigned short* h0g  = (unsigned short*)(ws + 6832128);              // 2048 B (pad)
  unsigned short* h1g  = (unsigned short*)(ws + 6834176);              // 2048 B (pad)
  float*        h1hist = (float*)(ws + 6836224);                       // 4194304 B
  unsigned*       bar  = (unsigned*)(ws + 11030528);                   // 8 B

  prep_w<<<NB, 256, 0, stream>>>(Wih0, Whh0, bih0, bhh0, Wih1, Whh1, bih1, bhh1, wsl, bias);
  prep_x<<<STEPS, 64, 0, stream>>>(batch, ln_w, ln_b, x_bf);
  init_state<<<1, 256, 0, stream>>>(h0g, h1g, bar);
  lstm_persist<<<NB, 256, 0, stream>>>(wsl, bias, x_bf, h0g, h1g, h1hist, bar);
  out_proj<<<STEPS, 256, 0, stream>>>(h1hist, fc_w, fc_b, out);
}
